// RelativeMultiHeadAttention_20194936226282
// MI455X (gfx1250) — matrix operand order repaired and checked by reading, not compiled, not run on hardware
//
#include <hip/hip_runtime.h>
#include <hip/hip_bf16.h>

// ---------------------------------------------------------------------------
// RelativeMultiHeadAttention forward for MI455X (gfx1250, wave32, WMMA + TDM).
// All matrix math runs through v_wmma_f32_16x16x32_f16 (f32 accumulate).
// Weight B-tiles are fetched by the Tensor Data Mover (tensor_load_to_lds,
// TENSORcnt); attn streaming (268 MB, > 192 MB L2) uses non-temporal hints.
// ---------------------------------------------------------------------------

typedef _Float16 v16h __attribute__((ext_vector_type(16)));
typedef float    v8f  __attribute__((ext_vector_type(8)));
typedef float    f32x4 __attribute__((ext_vector_type(4)));
typedef unsigned int u32x4 __attribute__((ext_vector_type(4)));
typedef int          i32x8 __attribute__((ext_vector_type(8)));
typedef int          i32x4 __attribute__((ext_vector_type(4)));

#define H_     8
#define DM_    512
#define DK_    64
#define MAXREL 16
#define NBINS  33
#define RPAD   36
#define B_     2
#define L_     2048

#if defined(__has_builtin)
#if __has_builtin(__builtin_amdgcn_tensor_load_to_lds) && \
    __has_builtin(__builtin_amdgcn_s_wait_tensorcnt)
#define USE_TDM 1
#endif
#endif
#ifndef USE_TDM
#define USE_TDM 0
#endif

union AFrag { v16h v; uint4 u[2]; };

// ---------------------------------------------------------------------------
// Wave-level 64x64 output tile, one 32-deep K step. 4 waves per 128-thread
// block; wave w owns quadrant (w>>1, w&1) = 2x2 WMMA 16x16 tiles.
// As: [64 rows][32 k] f16 row-major. Bs: [64 cols][32 k] f16 (col-major B).
// Fragment packing (this code's k order; A and B use the same one):
//   A: lane&15 = M row; halves h -> K = h + 16*(lane>=16), two 8-half runs (same as B).
//   B: lane&15 = N col; K = h + 16*(lane>=16) => one contiguous 16-half run.
//   C/D: M = vgpr + 8*(lane>=16), N = lane&15.
// ---------------------------------------------------------------------------
__device__ __forceinline__ void wave_mma_64x64x32(
    const _Float16* __restrict__ As,
    const _Float16* __restrict__ Bs,
    v8f acc[2][2])
{
  const int lane = threadIdx.x & 31;
  const int wave = threadIdx.x >> 5;
  const int wm = wave >> 1, wn = wave & 1;
  const int lo = lane & 15, hi = lane >> 4;

  AFrag a[2];
  v16h  b[2];
#pragma unroll
  for (int mt = 0; mt < 2; ++mt) {
    const int row = wm * 32 + mt * 16 + lo;
    a[mt].u[0] = *(const uint4*)(As + row * 32 + 16 * hi);       // K run 0
    a[mt].u[1] = *(const uint4*)(As + row * 32 + 16 * hi + 8);   // K run 1
  }
#pragma unroll
  for (int nt = 0; nt < 2; ++nt) {
    const int col = wn * 32 + nt * 16 + lo;
    b[nt] = *(const v16h*)(Bs + col * 32 + 16 * hi);             // 32B aligned
  }
#pragma unroll
  for (int mt = 0; mt < 2; ++mt)
#pragma unroll
    for (int nt = 0; nt < 2; ++nt)
      acc[mt][nt] = __builtin_amdgcn_wmma_f32_16x16x32_f16(
          false, a[mt].v, false, b[nt], (short)0, acc[mt][nt], false, false);
}

// 128 threads cooperatively stage a 64x32 tile into LDS.
__device__ __forceinline__ void load_tile_f32(_Float16* dst, const float* src, int ld)
{
  const int t   = threadIdx.x;      // 0..127
  const int row = t >> 1;
  const int c0  = (t & 1) * 16;
  const f32x4* s = (const f32x4*)(src + (size_t)row * ld + c0);
#pragma unroll
  for (int i = 0; i < 4; ++i) {
    f32x4 f = s[i];
    _Float16* d = dst + row * 32 + c0 + i * 4;
    d[0] = (_Float16)f.x; d[1] = (_Float16)f.y;
    d[2] = (_Float16)f.z; d[3] = (_Float16)f.w;
  }
}

// Same, but with non-temporal loads (streaming data that cannot live in L2).
__device__ __forceinline__ void load_tile_f32_nt(_Float16* dst, const float* src, int ld)
{
  const int t   = threadIdx.x;
  const int row = t >> 1;
  const int c0  = (t & 1) * 16;
  const f32x4* s = (const f32x4*)(src + (size_t)row * ld + c0);
#pragma unroll
  for (int i = 0; i < 4; ++i) {
    f32x4 f = __builtin_nontemporal_load(s + i);
    _Float16* d = dst + row * 32 + c0 + i * 4;
    d[0] = (_Float16)f.x; d[1] = (_Float16)f.y;
    d[2] = (_Float16)f.z; d[3] = (_Float16)f.w;
  }
}

__device__ __forceinline__ void load_tile_f16(_Float16* dst, const _Float16* src, int ld)
{
  const int t   = threadIdx.x;
  const int row = t >> 1;
  const int c0  = (t & 1) * 16;
  const uint4* s = (const uint4*)(src + (size_t)row * ld + c0);
  uint4*       d = (uint4*)(dst + row * 32 + c0);
  d[0] = s[0];
  d[1] = s[1];
}

#if USE_TDM
// ---------------------------------------------------------------------------
// TDM: DMA a 64-row x 32-half tile (row stride ldElems halves) into LDS.
// D# per CDNA5 ISA ch.8: group0 = {count=1, lds_addr, global_addr, type=2};
// group1 packs data_size=2B, tensor_dim0/1, tile_dim0=32/tile_dim1=64,
// tensor_dim0_stride (16-bit straddled fields). Groups 2/3 unused (2D tile).
// Issued by one wave; completion via s_wait_tensorcnt. Flat-address low 32
// bits of a __shared__ pointer are the LDS offset (ISA 10.2 aperture rules).
// ---------------------------------------------------------------------------
__device__ __forceinline__ void tdm_load_tile_64x32_f16(unsigned ldsAddr,
                                                        const _Float16* gsrc,
                                                        int ldElems)
{
  const unsigned long long ga = (unsigned long long)gsrc;
  u32x4 g0;
  g0.x = 1u;                                                   // count=1 (valid)
  g0.y = ldsAddr;                                              // lds_addr
  g0.z = (unsigned)ga;                                         // global_addr lo
  g0.w = (unsigned)((ga >> 32) & 0x01FFFFFFull) | (2u << 30);  // addr[56:32]|type=2
  i32x8 g1;
  g1[0] = 1 << 16;                                   // data_size=1 -> 2 bytes
  g1[1] = (int)(((unsigned)ldElems & 0xFFFFu) << 16);          // td0[15:0]
  g1[2] = (int)((((unsigned)ldElems >> 16) & 0xFFFFu)          // td0[31:16]
                | (0xFFFFu << 16));                            // td1[15:0]
  g1[3] = (int)(32u << 16);                          // td1[31:16]=0 | tile_dim0=32
  g1[4] = 64;                                        // tile_dim1=64, tile_dim2=0
  g1[5] = ldElems;                                   // tensor_dim0_stride[31:0]
  g1[6] = 0;                                         // stride0[47:32]|stride1 lo
  g1[7] = 0;
  i32x4 gz4 = {0, 0, 0, 0};
  i32x8 gz8 = {0, 0, 0, 0, 0, 0, 0, 0};
  __builtin_amdgcn_tensor_load_to_lds(g0, g1, gz4, gz4, gz8, 0);
}
#endif

// ---------------------------------------------------------------------------
// K0a: W [K][N] f32  ->  WT [N][K] f16  (so GEMM B-tiles are contiguous rows)
// ---------------------------------------------------------------------------
__global__ void transpose_cvt_kernel(const float* __restrict__ W,
                                     _Float16* __restrict__ WT)
{
  const int i = blockIdx.x * blockDim.x + threadIdx.x;   // 0 .. 512*512-1
  const int k = i >> 9;
  const int n = i & 511;
  WT[(size_t)n * DM_ + k] = (_Float16)W[(size_t)k * DM_ + n];
}

__global__ void cvt_kernel(const float* __restrict__ src,
                           _Float16* __restrict__ dst, int n)
{
  const int i = blockIdx.x * blockDim.x + threadIdx.x;
  if (i < n) dst[i] = (_Float16)src[i];
}

// ---------------------------------------------------------------------------
// K1: projection GEMM  C[4096,512] = X[4096,512] @ W + bias, WMMA tiled.
// VMODE==0: out = [bh][q][d] f16 (qp/kp).  VMODE==1: out = [bh][d][q] (vp^T).
// Weight tile via TDM (one wave issues; all waves sync on the block barrier).
// ---------------------------------------------------------------------------
template <int VMODE>
__global__ __launch_bounds__(128)
void proj_gemm_kernel(const float* __restrict__ X,
                      const _Float16* __restrict__ WT,   // [N][K]
                      const float* __restrict__ bias,
                      _Float16* __restrict__ out)
{
  __shared__ __align__(32) _Float16 As[64 * 32];
  __shared__ __align__(32) _Float16 Bs[64 * 32];
  const int m0 = blockIdx.x * 64;
  const int n0 = blockIdx.y * 64;
  v8f acc[2][2] = {};
  for (int k0 = 0; k0 < DM_; k0 += 32) {
#if USE_TDM
    if (threadIdx.x < 32)
      tdm_load_tile_64x32_f16((unsigned)(unsigned long long)Bs,
                              WT + (size_t)n0 * DM_ + k0, DM_);
#else
    load_tile_f16(Bs, WT + (size_t)n0 * DM_ + k0, DM_);
#endif
    load_tile_f32(As, X + (size_t)m0 * DM_ + k0, DM_);
#if USE_TDM
    if (threadIdx.x < 32) __builtin_amdgcn_s_wait_tensorcnt(0);
#endif
    __syncthreads();
    wave_mma_64x64x32(As, Bs, acc);
    __syncthreads();
  }
  const int lane = threadIdx.x & 31, wave = threadIdx.x >> 5;
  const int wm = wave >> 1, wn = wave & 1;
  const int lo = lane & 15, hi = lane >> 4;
  const int head = n0 / DK_;                 // BN==DK==64, head-aligned
#pragma unroll
  for (int mt = 0; mt < 2; ++mt)
#pragma unroll
    for (int nt = 0; nt < 2; ++nt)
#pragma unroll
      for (int v = 0; v < 8; ++v) {
        const int m = m0 + wm * 32 + mt * 16 + hi * 8 + v;   // b*L + q
        const int n = n0 + wn * 32 + nt * 16 + lo;
        const int b = m / L_, q = m % L_;
        const int d = n - head * DK_;
        const float val = acc[mt][nt][v] + bias[n];
        if (VMODE)
          out[((size_t)(b * H_ + head) * DK_ + d) * L_ + q] = (_Float16)val;
        else
          out[((size_t)(b * H_ + head) * L_ + q) * DK_ + d] = (_Float16)val;
      }
}

// ---------------------------------------------------------------------------
// K2: r[row][j] = qp[row,:] . rpe_k[j,:]   (row = bh*L+q, j = 0..32)
// Tiny (69M MACs); everything is L2-resident.
// ---------------------------------------------------------------------------
__global__ void rel_dot_kernel(const _Float16* __restrict__ qp,
                               const _Float16* __restrict__ rpe,
                               float* __restrict__ r)
{
  const int row = blockIdx.x;
  const int j   = threadIdx.x;
  if (j < NBINS) {
    const _Float16* qr = qp + (size_t)row * DK_;
    const _Float16* rr = rpe + (size_t)j * DK_;
    float s = 0.f;
#pragma unroll
    for (int d = 0; d < DK_; ++d) s += (float)qr[d] * (float)rr[d];
    r[(size_t)row * RPAD + j] = s;
  }
}

// ---------------------------------------------------------------------------
// K3: scores = (qp @ kp^T + rel)/8, masked; written f32 into the attn output.
// kp is [bh][k][d] row-major == exactly the col-major B tile we need.
// ---------------------------------------------------------------------------
__global__ __launch_bounds__(128)
void score_kernel(const _Float16* __restrict__ qp,
                  const _Float16* __restrict__ kp,
                  const float* __restrict__ r,
                  const int* __restrict__ mask,
                  float* __restrict__ attn)
{
  __shared__ __align__(32) _Float16 As[64 * 32];
  __shared__ __align__(32) _Float16 Bs[64 * 32];
  const int m0 = blockIdx.x * 64;     // q tile
  const int n0 = blockIdx.y * 64;     // k tile
  const int bh = blockIdx.z;
  const int b  = bh / H_;
  const _Float16* qbase = qp + (size_t)bh * L_ * DK_;
  const _Float16* kbase = kp + (size_t)bh * L_ * DK_;
  v8f acc[2][2] = {};
#pragma unroll
  for (int k0 = 0; k0 < DK_; k0 += 32) {
    load_tile_f16(As, qbase + (size_t)m0 * DK_ + k0, DK_);
    load_tile_f16(Bs, kbase + (size_t)n0 * DK_ + k0, DK_);
    __syncthreads();
    wave_mma_64x64x32(As, Bs, acc);
    __syncthreads();
  }
  const int lane = threadIdx.x & 31, wave = threadIdx.x >> 5;
  const int wm = wave >> 1, wn = wave & 1;
  const int lo = lane & 15, hi = lane >> 4;
#pragma unroll
  for (int mt = 0; mt < 2; ++mt)
#pragma unroll
    for (int nt = 0; nt < 2; ++nt)
#pragma unroll
      for (int v = 0; v < 8; ++v) {
        const int q  = m0 + wm * 32 + mt * 16 + hi * 8 + v;
        const int kc = n0 + wn * 32 + nt * 16 + lo;
        int delta = kc - q;
        delta = min(max(delta, -MAXREL), MAXREL);
        float s = acc[mt][nt][v]
                + r[((size_t)bh * L_ + q) * RPAD + (delta + MAXREL)];
        s *= 0.125f;                                   // 1/sqrt(64)
        if (mask[((size_t)b * L_ + q) * L_ + kc] == 0) s = -1e9f;
        attn[((size_t)bh * L_ + q) * L_ + kc] = s;
      }
}

// ---------------------------------------------------------------------------
// K4: in-place row softmax over attn rows of length L (256 thr, 8 elem/thr).
// Final store is NT: 268 MB stream, consumed exactly once by the AV GEMM.
// ---------------------------------------------------------------------------
__global__ __launch_bounds__(256)
void softmax_kernel(float* __restrict__ attn)
{
  __shared__ float red[8];
  const size_t base = (size_t)blockIdx.x * L_;
  const int t = threadIdx.x;
  float v[8];
  float m = -3.0e38f;
#pragma unroll
  for (int i = 0; i < 8; ++i) {
    v[i] = attn[base + t + i * 256];
    m = fmaxf(m, v[i]);
  }
#pragma unroll
  for (int off = 16; off > 0; off >>= 1) m = fmaxf(m, __shfl_xor(m, off, 32));
  if ((t & 31) == 0) red[t >> 5] = m;
  __syncthreads();
  float bm = red[0];
#pragma unroll
  for (int w = 1; w < 8; ++w) bm = fmaxf(bm, red[w]);
  float s = 0.f;
#pragma unroll
  for (int i = 0; i < 8; ++i) { v[i] = __expf(v[i] - bm); s += v[i]; }
#pragma unroll
  for (int off = 16; off > 0; off >>= 1) s += __shfl_xor(s, off, 32);
  __syncthreads();
  if ((t & 31) == 0) red[t >> 5] = s;
  __syncthreads();
  float bs = 0.f;
#pragma unroll
  for (int w = 0; w < 8; ++w) bs += red[w];
  const float inv = 1.0f / bs;
#pragma unroll
  for (int i = 0; i < 8; ++i)
    __builtin_nontemporal_store(v[i] * inv, &attn[base + t + i * 256]);
}

// ---------------------------------------------------------------------------
// K5: out = attn @ vp  per (b,h): [2048 x 2048] x [2048 x 64], K-streamed.
// vp stored transposed [bh][d][L] so B tiles are contiguous row copies.
// attn loads are NT (read-once stream) with a prefetch on the next K tile.
// ---------------------------------------------------------------------------
__global__ __launch_bounds__(128)
void av_gemm_kernel(const float* __restrict__ attn,
                    const _Float16* __restrict__ vpT,
                    _Float16* __restrict__ concat)
{
  __shared__ __align__(32) _Float16 As[64 * 32];
  __shared__ __align__(32) _Float16 Bs[64 * 32];
  const int m0 = blockIdx.x * 64;      // q tile; N = full 64 (= d_k)
  const int bh = blockIdx.y;
  const int b  = bh / H_, h = bh % H_;
  const float*    abase = attn + (size_t)bh * L_ * L_;
  const _Float16* vbase = vpT  + (size_t)bh * DK_ * L_;
  v8f acc[2][2] = {};
  for (int k0 = 0; k0 < L_; k0 += 32) {
    if (k0 + 32 < L_)   // prefetch next attn tile (global_prefetch_b8)
      __builtin_prefetch(abase + (size_t)(m0 + (threadIdx.x >> 1)) * L_ +
                             (k0 + 32) + (threadIdx.x & 1) * 16, 0, 0);
    load_tile_f32_nt(As, abase + (size_t)m0 * L_ + k0, L_);
    load_tile_f16(Bs, vbase + k0, L_);
    __syncthreads();
    wave_mma_64x64x32(As, Bs, acc);
    __syncthreads();
  }
  const int lane = threadIdx.x & 31, wave = threadIdx.x >> 5;
  const int wm = wave >> 1, wn = wave & 1;
  const int lo = lane & 15, hi = lane >> 4;
#pragma unroll
  for (int mt = 0; mt < 2; ++mt)
#pragma unroll
    for (int nt = 0; nt < 2; ++nt)
#pragma unroll
      for (int v = 0; v < 8; ++v) {
        const int q = m0 + wm * 32 + mt * 16 + hi * 8 + v;
        const int d = wn * 32 + nt * 16 + lo;
        concat[((size_t)b * L_ + q) * DM_ + h * DK_ + d] =
            (_Float16)acc[mt][nt][v];
      }
}

// ---------------------------------------------------------------------------
// K6: final = concat @ Wo + bo -> f32 output (never re-read: NT stores).
// Weight tile via TDM as in K1.
// ---------------------------------------------------------------------------
__global__ __launch_bounds__(128)
void out_gemm_kernel(const _Float16* __restrict__ Xc,   // [4096,512] f16
                     const _Float16* __restrict__ WT,   // [N][K]
                     const float* __restrict__ bias,
                     float* __restrict__ out)           // [4096,512] f32
{
  __shared__ __align__(32) _Float16 As[64 * 32];
  __shared__ __align__(32) _Float16 Bs[64 * 32];
  const int m0 = blockIdx.x * 64;
  const int n0 = blockIdx.y * 64;
  v8f acc[2][2] = {};
  for (int k0 = 0; k0 < DM_; k0 += 32) {
#if USE_TDM
    if (threadIdx.x < 32)
      tdm_load_tile_64x32_f16((unsigned)(unsigned long long)Bs,
                              WT + (size_t)n0 * DM_ + k0, DM_);
#else
    load_tile_f16(Bs, WT + (size_t)n0 * DM_ + k0, DM_);
#endif
    load_tile_f16(As, Xc + (size_t)m0 * DM_ + k0, DM_);
#if USE_TDM
    if (threadIdx.x < 32) __builtin_amdgcn_s_wait_tensorcnt(0);
#endif
    __syncthreads();
    wave_mma_64x64x32(As, Bs, acc);
    __syncthreads();
  }
  const int lane = threadIdx.x & 31, wave = threadIdx.x >> 5;
  const int wm = wave >> 1, wn = wave & 1;
  const int lo = lane & 15, hi = lane >> 4;
#pragma unroll
  for (int mt = 0; mt < 2; ++mt)
#pragma unroll
    for (int nt = 0; nt < 2; ++nt)
#pragma unroll
      for (int v = 0; v < 8; ++v) {
        const int m = m0 + wm * 32 + mt * 16 + hi * 8 + v;
        const int n = n0 + wn * 32 + nt * 16 + lo;
        __builtin_nontemporal_store(acc[mt][nt][v] + bias[n],
                                    &out[(size_t)m * DM_ + n]);
      }
}

// ---------------------------------------------------------------------------
extern "C" void kernel_launch(void* const* d_in, const int* in_sizes, int n_in,
                              void* d_out, int out_size, void* d_ws, size_t ws_size,
                              hipStream_t stream)
{
  (void)in_sizes; (void)n_in; (void)out_size; (void)ws_size;

  const float* q    = (const float*)d_in[0];
  const float* k    = (const float*)d_in[1];
  const float* v    = (const float*)d_in[2];
  const int*   mask = (const int*)  d_in[3];
  const float* Wq   = (const float*)d_in[4];
  const float* bq   = (const float*)d_in[5];
  const float* Wk   = (const float*)d_in[6];
  const float* bk   = (const float*)d_in[7];
  const float* Wv   = (const float*)d_in[8];
  const float* bv   = (const float*)d_in[9];
  const float* Wo   = (const float*)d_in[10];
  const float* bo   = (const float*)d_in[11];
  const float* rpe  = (const float*)d_in[12];

  float* out_final = (float*)d_out;                       // [2,2048,512]
  float* out_attn  = out_final + (size_t)B_ * L_ * DM_;   // [2,8,2048,2048]

  char*  ws  = (char*)d_ws;
  size_t off = 0;
  auto alloc = [&](size_t bytes) {
    size_t o = off;
    off = (off + bytes + 255) & ~(size_t)255;
    return o;
  };
  const size_t wBytes = (size_t)DM_ * DM_ * sizeof(_Float16);
  const size_t hBytes = (size_t)B_ * H_ * L_ * DK_ * sizeof(_Float16);
  _Float16* WqT   = (_Float16*)(ws + alloc(wBytes));
  _Float16* WkT   = (_Float16*)(ws + alloc(wBytes));
  _Float16* WvT   = (_Float16*)(ws + alloc(wBytes));
  _Float16* WoT   = (_Float16*)(ws + alloc(wBytes));
  _Float16* rpe16 = (_Float16*)(ws + alloc((size_t)NBINS * DK_ * sizeof(_Float16)));
  _Float16* qp16  = (_Float16*)(ws + alloc(hBytes));
  _Float16* kp16  = (_Float16*)(ws + alloc(hBytes));
  _Float16* vpT16 = (_Float16*)(ws + alloc(hBytes));
  float*    rbuf  = (float*)   (ws + alloc((size_t)B_ * H_ * L_ * RPAD * sizeof(float)));
  _Float16* cc16  = (_Float16*)(ws + alloc((size_t)B_ * L_ * DM_ * sizeof(_Float16)));

  // weight transpose/convert
  transpose_cvt_kernel<<<(DM_ * DM_) / 256, 256, 0, stream>>>(Wq, WqT);
  transpose_cvt_kernel<<<(DM_ * DM_) / 256, 256, 0, stream>>>(Wk, WkT);
  transpose_cvt_kernel<<<(DM_ * DM_) / 256, 256, 0, stream>>>(Wv, WvT);
  transpose_cvt_kernel<<<(DM_ * DM_) / 256, 256, 0, stream>>>(Wo, WoT);
  cvt_kernel<<<(NBINS * DK_ + 255) / 256, 256, 0, stream>>>(rpe, rpe16, NBINS * DK_);

  // projections (M=4096, N=512)
  dim3 gProj(64, 8);
  proj_gemm_kernel<0><<<gProj, 128, 0, stream>>>(q, WqT, bq, qp16);
  proj_gemm_kernel<0><<<gProj, 128, 0, stream>>>(k, WkT, bk, kp16);
  proj_gemm_kernel<1><<<gProj, 128, 0, stream>>>(v, WvT, bv, vpT16);

  // relative-position dot table
  rel_dot_kernel<<<B_ * H_ * L_, 64, 0, stream>>>(qp16, rpe16, rbuf);

  // scores -> attn (raw), softmax in place
  score_kernel<<<dim3(L_ / 64, L_ / 64, B_ * H_), 128, 0, stream>>>(
      qp16, kp16, rbuf, mask, out_attn);
  softmax_kernel<<<B_ * H_ * L_, 256, 0, stream>>>(out_attn);

  // out = attn @ vp ; final = concat @ Wo + bo
  av_gemm_kernel<<<dim3(L_ / 64, B_ * H_), 128, 0, stream>>>(out_attn, vpT16, cc16);
  out_gemm_kernel<<<dim3(64, 8), 128, 0, stream>>>(cc16, WoT, bo, out_final);
}
